// Actor_77171972374917
// MI455X (gfx1250) — matrix operand order repaired and checked by reading, not compiled, not run on hardware
//
#include <hip/hip_runtime.h>

#define GG 512
#define NN 256
#define EE 128

typedef __attribute__((ext_vector_type(16))) __bf16 v16bf;
typedef __attribute__((ext_vector_type(8)))  float  v8f;

// ---------------------------------------------------------------------------
// Low-level helpers
// ---------------------------------------------------------------------------

__device__ __forceinline__ unsigned lds_off(const void* p) {
  // On gfx1250 the low 32 bits of a generic LDS address are the LDS byte offset.
  return (unsigned)(unsigned long long)p;
}

__device__ __forceinline__ void async_b128(unsigned lds, unsigned long long base, unsigned off) {
  asm volatile("global_load_async_to_lds_b128 %0, %1, %2"
               :: "v"(lds), "v"(off), "s"(base) : "memory");
}
__device__ __forceinline__ void async_b64(unsigned lds, unsigned long long base, unsigned off) {
  asm volatile("global_load_async_to_lds_b64 %0, %1, %2"
               :: "v"(lds), "v"(off), "s"(base) : "memory");
}
__device__ __forceinline__ void wait_async0() { asm volatile("s_wait_asynccnt 0" ::: "memory"); }
__device__ __forceinline__ void wait_async3() { asm volatile("s_wait_asynccnt 3" ::: "memory"); }

// A-matrix 16x32 bf16 fragment from a row-major LDS tile (rowStride in elems).
// Lane L: M = L%16 ; lanes 0-15 hold K in {0..7,16..23}, lanes 16-31 K+8.
__device__ __forceinline__ v16bf load_a_frag_lds(const __bf16* base, int rowStride, int k0) {
  union { v16bf v; unsigned u[8]; } f;
  int lane = threadIdx.x & 31;
  int m    = lane & 15;
  int hi   = (lane >> 4) & 1;
  const __bf16* row = base + m * rowStride + k0 + hi * 8;
#pragma unroll
  for (int e2 = 0; e2 < 8; ++e2) {
    int K = 2 * e2 + ((e2 >= 4) ? 8 : 0);
    f.u[e2] = *(const unsigned*)(row + K);
  }
  return f.v;
}

// A-matrix 16x32 bf16 fragment converted on the fly from an int32 adjacency
// chunk in LDS, laid out [16 rows][32 cols].
__device__ __forceinline__ v16bf a_frag_from_adj(const int* adjChunk) {
  v16bf v;
  int lane = threadIdx.x & 31;
  int m    = lane & 15;
  int hi   = (lane >> 4) & 1;
  const int* row = adjChunk + m * 32 + hi * 8;
#pragma unroll
  for (int e = 0; e < 16; ++e) {
    int K = e + ((e >= 8) ? 8 : 0);
    v[e] = (__bf16)(float)row[K];
  }
  return v;
}

// B-matrix 32x16 bf16 fragment from a row-major [K][N] bf16 tile.
// Lane L: N = c0 + L%16 ; element e -> K = k0 + e + (e>=8)*8 + (L>=16)*8.
__device__ __forceinline__ v16bf load_b_frag_rm(const __bf16* base, int rowStride, int k0, int c0) {
  v16bf v;
  int lane = threadIdx.x & 31;
  int n    = c0 + (lane & 15);
  int hi   = (lane >> 4) & 1;
#pragma unroll
  for (int e = 0; e < 16; ++e) {
    int K = k0 + e + ((e >= 8) ? 8 : 0) + hi * 8;
    v[e] = base[K * rowStride + n];
  }
  return v;
}

// B-matrix 32x16 bf16 fragment from a K-contiguous transposed weight wT[N][K].
__device__ __forceinline__ v16bf load_b_frag_t(const __bf16* baseT, int kStride, int c0, int k0) {
  union { v16bf v; unsigned u[8]; } f;
  int lane = threadIdx.x & 31;
  int n    = c0 + (lane & 15);
  int hi   = (lane >> 4) & 1;
  const __bf16* p = baseT + n * kStride + k0 + hi * 8;
#pragma unroll
  for (int e2 = 0; e2 < 8; ++e2) f.u[e2] = *(const unsigned*)(p + 2 * e2 + ((e2 >= 4) ? 8 : 0));
  return f.v;
}

// C/D 16x16 f32 accumulator -> bf16 LDS tile. Lane L: N=c0+L%16, M=r+(L>=16)*8.
__device__ __forceinline__ void store_c_bf16(__bf16* base, int rowStride, int c0, v8f acc, bool relu) {
  int lane = threadIdx.x & 31;
  int n    = c0 + (lane & 15);
  int hiM  = (lane >> 4) ? 8 : 0;
#pragma unroll
  for (int r = 0; r < 8; ++r) {
    float v = acc[r];
    if (relu) v = fmaxf(v, 0.f);
    base[(r + hiM) * rowStride + n] = (__bf16)v;
  }
}

// Shared final linear: D = relu(src(16x128) @ wT + bias) written to global bf16.
__device__ __forceinline__ void mlp_tile_to_global(const __bf16* src /*16x136 LDS*/,
                                                   const __bf16* wT, const float* bias,
                                                   __bf16* dstBase, int m0) {
  int wave = threadIdx.x >> 5;
  int lane = threadIdx.x & 31;
  v8f acc = {};
#pragma unroll
  for (int k0 = 0; k0 < EE; k0 += 32) {
    v16bf a = load_a_frag_lds(src, 136, k0);
    v16bf b = load_b_frag_t(wT, EE, wave * 16, k0);
    acc = __builtin_amdgcn_wmma_f32_16x16x32_bf16(false, a, false, b, (short)0, acc, false, false);
  }
  int   n   = wave * 16 + (lane & 15);
  int   hiM = (lane >> 4) ? 8 : 0;
  float bb  = bias[n];
#pragma unroll
  for (int r = 0; r < 8; ++r) {
    float v = fmaxf(acc[r] + bb, 0.f);
    dstBase[(size_t)(m0 + r + hiM) * EE + n] = (__bf16)v;
  }
}

__device__ __forceinline__ float rnd01(unsigned g, unsigned which) {
  unsigned long long z = ((unsigned long long)g << 32) ^
                         (0x9E3779B97F4A7C15ull * (which + 1)) ^ 42ull;
  z += 0x9E3779B97F4A7C15ull;
  z = (z ^ (z >> 30)) * 0xBF58476D1CE4E5B9ull;
  z = (z ^ (z >> 27)) * 0x94D049BB133111EBull;
  z ^= z >> 31;
  return (float)(z >> 40) * (1.0f / 16777216.0f);
}

// ---------------------------------------------------------------------------
// k_prep: transpose-convert conv weights to bf16 wT[N][K]; conv1 W1 colsums.
//   blocks 0..4: {c1W2, c2W1, c2W2, c3W1, c3W2} ; block 5: cs1
// ---------------------------------------------------------------------------
__global__ __launch_bounds__(256) void k_prep(const float* c1W1, const float* c1W2,
                                              const float* c2W1, const float* c2W2,
                                              const float* c3W1, const float* c3W2,
                                              __bf16* wT, float* cs1) {
  int b = blockIdx.x, t = threadIdx.x;
  const float* srcs[5] = {c1W2, c2W1, c2W2, c3W1, c3W2};
  if (b < 5) {
    const float* src = srcs[b];
    __bf16* dst = wT + (size_t)b * EE * EE;
    for (int i = t; i < EE * EE; i += 256) {
      int n = i / EE, k = i % EE;
      dst[n * EE + k] = (__bf16)src[k * EE + n];
    }
  } else {
    for (int o = t; o < EE; o += 256) {
      float s = 0.f;
      for (int f = 0; f < 32; ++f) s += c1W1[f * EE + o];
      cs1[o] = s;
    }
  }
}

// ---------------------------------------------------------------------------
// k_layer1: exploits x0 == ones.  s = 1 + deg  (broadcast over 32 feats)
//   t = relu((1+deg)*cs1 + b1);  h = relu(t @ W2 + b2)
// Also emits deg[] for the heads.  grid = (N/16, G), 256 threads.
// ---------------------------------------------------------------------------
__global__ __launch_bounds__(256) void k_layer1(const int* __restrict__ adj,
                                                const float* __restrict__ cs1,
                                                const float* __restrict__ b1,
                                                const __bf16* __restrict__ w2T,
                                                const float* __restrict__ b2,
                                                __bf16* __restrict__ xout,
                                                float* __restrict__ degOut) {
  __shared__ int    adjrows[16 * NN];   // 16 KB
  __shared__ int    degI[16];
  __shared__ __bf16 ts[16 * 136];

  const int g  = blockIdx.y;
  const int m0 = blockIdx.x * 16;
  const int t  = threadIdx.x;

  if (t < 16) degI[t] = 0;

  // Async-stage 16 contiguous adjacency rows (16 KB) into LDS.
  unsigned           ldsA = lds_off(adjrows);
  unsigned long long gA   = (unsigned long long)(adj + ((size_t)g * NN + m0) * NN);
#pragma unroll
  for (int i = 0; i < 4; ++i)
    async_b128(ldsA + (unsigned)(t * 16 + i * 4096), gA, (unsigned)(t * 16 + i * 4096));
  wait_async0();
  __syncthreads();

  // Row sums -> degrees. 16 partials of 16 elements per row.
  {
    int r = t & 15, seg = t >> 4;
    int s = 0;
#pragma unroll
    for (int j = 0; j < 16; ++j) s += adjrows[r * NN + seg * 16 + j];
    atomicAdd(&degI[r], s);
  }
  __syncthreads();

  if (t < 16) degOut[(size_t)g * NN + m0 + t] = (float)degI[t];

  // t-tile = relu((1+deg) * cs1 + b1) in bf16
  for (int i = t; i < 16 * EE; i += 256) {
    int r = i / EE, o = i % EE;
    float v = (1.f + (float)degI[r]) * cs1[o] + b1[o];
    ts[r * 136 + o] = (__bf16)fmaxf(v, 0.f);
  }
  __syncthreads();

  mlp_tile_to_global(ts, w2T, b2, xout + (size_t)g * NN * EE, m0);
}

// ---------------------------------------------------------------------------
// k_layer (conv2 / conv3): h = relu(MLP2(x + adj@x)).  grid = (N/16, G).
// Double-buffered async pipeline over 32-node chunks; 8 waves x 16-col strips.
// ---------------------------------------------------------------------------
__global__ __launch_bounds__(256) void k_layer(const int* __restrict__ adj,
                                               const __bf16* __restrict__ xin,
                                               const __bf16* __restrict__ w1T,
                                               const float* __restrict__ b1,
                                               const __bf16* __restrict__ w2T,
                                               const float* __restrict__ b2,
                                               __bf16* __restrict__ xout) {
  __shared__ int    adjbuf[2][16 * 32];   // 2 x 2 KB
  __shared__ __bf16 xbuf[2][32 * EE];     // 2 x 8 KB
  __shared__ __bf16 ss[16 * 136];
  __shared__ __bf16 ts[16 * 136];

  const int g    = blockIdx.y;
  const int m0   = blockIdx.x * 16;
  const int t    = threadIdx.x;
  const int wave = t >> 5;
  const int lane = t & 31;

  const unsigned long long adjBase = (unsigned long long)(adj + (size_t)g * NN * NN);
  const unsigned long long xBase   = (unsigned long long)(xin + (size_t)g * NN * EE);

  auto issue_chunk = [&](int k0, int buf) {
    // x chunk: 32 nodes x 128 bf16 feats = 8192 B, contiguous.
    unsigned ldsX = lds_off(&xbuf[buf][0]);
    unsigned gOfX = (unsigned)(k0 * EE * 2);
    async_b128(ldsX + (unsigned)(t * 16), xBase, gOfX + (unsigned)(t * 16));
    async_b128(ldsX + (unsigned)(t * 16 + 4096), xBase, gOfX + (unsigned)(t * 16 + 4096));
    // adjacency chunk: 16 rows x 32 int32 (8 B per thread, all 256 threads).
    unsigned ldsA = lds_off(&adjbuf[buf][0]);
    int row = t >> 4, c8 = t & 15;
    async_b64(ldsA + (unsigned)(row * 128 + c8 * 8), adjBase,
              (unsigned)(((m0 + row) * NN + k0) * 4 + c8 * 8));
  };

  // ---- aggregation: agg = adj[m0:m0+16, :] @ x  (K loop over 256 nodes) ----
  v8f acc = {};
  issue_chunk(0, 0);
  for (int c = 0; c < 8; ++c) {
    int buf = c & 1;
    if (c < 7) { issue_chunk((c + 1) * 32, buf ^ 1); wait_async3(); }
    else       { wait_async0(); }
    __syncthreads();
    v16bf a = a_frag_from_adj(&adjbuf[buf][0]);
    v16bf b = load_b_frag_rm(&xbuf[buf][0], EE, 0, wave * 16);
    acc = __builtin_amdgcn_wmma_f32_16x16x32_bf16(false, a, false, b, (short)0, acc, false, false);
    __syncthreads();
  }

  // s = x + agg   (GIN eps=0), kept as bf16 in LDS
  {
    int n   = wave * 16 + (lane & 15);
    int hiM = (lane >> 4) ? 8 : 0;
#pragma unroll
    for (int r = 0; r < 8; ++r)
      acc[r] += (float)xin[(size_t)g * NN * EE + (size_t)(m0 + r + hiM) * EE + n];
    store_c_bf16(ss, 136, wave * 16, acc, false);
  }
  __syncthreads();

  // ---- hidden = relu(s @ W1 + b1) ----
  {
    v8f acc1 = {};
#pragma unroll
    for (int k0 = 0; k0 < EE; k0 += 32) {
      v16bf a = load_a_frag_lds(ss, 136, k0);
      v16bf b = load_b_frag_t(w1T, EE, wave * 16, k0);
      acc1 = __builtin_amdgcn_wmma_f32_16x16x32_bf16(false, a, false, b, (short)0, acc1, false, false);
    }
    int   n  = wave * 16 + (lane & 15);
    float bb = b1[n];
#pragma unroll
    for (int r = 0; r < 8; ++r) acc1[r] += bb;
    store_c_bf16(ts, 136, wave * 16, acc1, true);
  }
  __syncthreads();

  // ---- out = relu(hidden @ W2 + b2) -> global bf16 ----
  mlp_tile_to_global(ts, w2T, b2, xout + (size_t)g * NN * EE, m0);
}

// ---------------------------------------------------------------------------
// k_heads: masked softmax heads + deterministic categorical sampling.
// One block (256 threads) per graph; dynamic LDS.
// ---------------------------------------------------------------------------
__global__ __launch_bounds__(256) void k_heads(const __bf16* __restrict__ x,
                                               const float* __restrict__ deg,
                                               const float* wA1, const float* bA1,
                                               const float* wA2, const float* bA2,
                                               const float* wB1, const float* bB1,
                                               const float* wB2, const float* bB2,
                                               const float* wC1, const float* bC1,
                                               const float* wC2, const float* bC2,
                                               float* __restrict__ pA, float* __restrict__ pB,
                                               float* __restrict__ pC, float* __restrict__ act) {
  extern __shared__ char smem[];
  __bf16* xL    = (__bf16*)smem;                    // 256*128 bf16 = 65536 B
  float*  wA1L  = (float*)(smem + 65536);           // 128*32
  float*  wB1L  = wA1L + 128 * 32;                  // 256*32
  float*  wC1L  = wB1L + 256 * 32;                  // 128*32
  float*  red   = wC1L + 128 * 32;                  // 256
  float*  eArr  = red + 256;                        // 256
  float*  gemb  = eArr + 256;                       // 128
  float*  vflag = gemb + 128;                       // 256
  __shared__ int   sInt[4];   // count, lastValid, first, second
  __shared__ float sFlt[4];

  const int g = blockIdx.x;
  const int t = threadIdx.x;

  if (t == 0) { sInt[0] = 0; sInt[1] = -1; }

  { // cooperative loads
    const uint4* src = (const uint4*)(x + (size_t)g * NN * EE);
    uint4* dst = (uint4*)xL;
    for (int i = t; i < NN * EE / 8; i += 256) dst[i] = src[i];
    for (int i = t; i < 128 * 32; i += 256) wA1L[i] = wA1[i];
    for (int i = t; i < 256 * 32; i += 256) wB1L[i] = wB1[i];
    for (int i = t; i < 128 * 32; i += 256) wC1L[i] = wC1[i];
  }
  __syncthreads();

  // validity
  int v = (deg[(size_t)g * NN + t] > 0.f) ? 1 : 0;
  vflag[t] = (float)v;
  atomicAdd(&sInt[0], v);
  atomicMax(&sInt[1], v ? t : -1);
  __syncthreads();
  const int count = sInt[0];
  const int last  = sInt[1];

  // ---------- head A (non-scaffold mask) ----------
  float lA = bA2[0];
  for (int o = 0; o < 32; ++o) {
    float h = bA1[o];
    for (int f = 0; f < EE; ++f) h += (float)xL[t * EE + f] * wA1L[f * 32 + o];
    lA += fmaxf(h, 0.f) * wA2[o];
  }
  int maskA = v && (t != last);
  red[t] = maskA ? lA : -3.0e38f;
  __syncthreads();
  for (int s = 128; s > 0; s >>= 1) { if (t < s) red[t] = fmaxf(red[t], red[t + s]); __syncthreads(); }
  float mA = red[0];
  __syncthreads();
  float eA = maskA ? __expf(lA - mA) : 0.f;
  eArr[t] = eA; red[t] = eA;
  __syncthreads();
  for (int s = 128; s > 0; s >>= 1) { if (t < s) red[t] += red[t + s]; __syncthreads(); }
  float dA = red[0];
  pA[(size_t)g * NN + t] = eA / dA;
  if (t == 0) {
    float u = rnd01(g, 0) * dA, cs = 0.f;
    int sel = last;
    for (int i = 0; i < NN; ++i) { cs += eArr[i]; if (cs >= u && eArr[i] > 0.f) { sel = i; break; } }
    sInt[2] = sel;
  }
  __syncthreads();
  const int first = sInt[2];

  // ---------- head B (concat with first-node embedding; valid mask) ----------
  float lB = bB2[0];
  for (int o = 0; o < 32; ++o) {
    float h = bB1[o];
    for (int f = 0; f < EE; ++f) h += (float)xL[t * EE + f] * wB1L[f * 32 + o];
    for (int f = 0; f < EE; ++f) h += (float)xL[first * EE + f] * wB1L[(EE + f) * 32 + o];
    lB += fmaxf(h, 0.f) * wB2[o];
  }
  red[t] = v ? lB : -3.0e38f;
  __syncthreads();
  for (int s = 128; s > 0; s >>= 1) { if (t < s) red[t] = fmaxf(red[t], red[t + s]); __syncthreads(); }
  float mB = red[0];
  __syncthreads();
  float eB = v ? __expf(lB - mB) : 0.f;
  eArr[t] = eB; red[t] = eB;
  __syncthreads();
  for (int s = 128; s > 0; s >>= 1) { if (t < s) red[t] += red[t + s]; __syncthreads(); }
  float dB = red[0];
  pB[(size_t)g * NN + t] = eB / dB;
  if (t == 0) {
    float u = rnd01(g, 1) * dB, cs = 0.f;
    int sel = last;
    for (int i = 0; i < NN; ++i) { cs += eArr[i]; if (cs >= u && eArr[i] > 0.f) { sel = i; break; } }
    sInt[3] = sel;
  }

  // ---------- head C (global mean pool) ----------
  if (t < EE) {
    float s = 0.f;
    for (int n = 0; n < NN; ++n) s += vflag[n] * (float)xL[n * EE + t];
    gemb[t] = s / (float)count;
  }
  __syncthreads();
  if (t < 32) {
    float h = bC1[t];
    for (int f = 0; f < EE; ++f) h += gemb[f] * wC1L[f * 32 + t];
    red[t] = fmaxf(h, 0.f);
  }
  __syncthreads();
  if (t < 2) {
    float z = bC2[t];
    for (int o = 0; o < 32; ++o) z += red[o] * wC2[o * 2 + t];
    sFlt[t] = z;
  }
  __syncthreads();
  if (t == 0) {
    float z0 = sFlt[0], z1 = sFlt[1];
    float mz = fmaxf(z0, z1);
    float e0 = __expf(z0 - mz), e1 = __expf(z1 - mz);
    float d  = e0 + e1;
    pC[(size_t)g * 2 + 0] = e0 / d;
    pC[(size_t)g * 2 + 1] = e1 / d;
    int isEnd = (rnd01(g, 2) < e0 / d) ? 0 : 1;
    act[(size_t)g * 3 + 0] = (float)sInt[2];
    act[(size_t)g * 3 + 1] = (float)sInt[3];
    act[(size_t)g * 3 + 2] = (float)isEnd;
  }
}

// ---------------------------------------------------------------------------
extern "C" void kernel_launch(void* const* d_in, const int* in_sizes, int n_in,
                              void* d_out, int out_size, void* d_ws, size_t ws_size,
                              hipStream_t stream) {
  (void)in_sizes; (void)n_in; (void)out_size; (void)ws_size;
  const int*   states = (const int*)d_in[0];
  const float* c1W1 = (const float*)d_in[1],  *c1b1 = (const float*)d_in[2];
  const float* c1W2 = (const float*)d_in[3],  *c1b2 = (const float*)d_in[4];
  const float* c2W1 = (const float*)d_in[5],  *c2b1 = (const float*)d_in[6];
  const float* c2W2 = (const float*)d_in[7],  *c2b2 = (const float*)d_in[8];
  const float* c3W1 = (const float*)d_in[9],  *c3b1 = (const float*)d_in[10];
  const float* c3W2 = (const float*)d_in[11], *c3b2 = (const float*)d_in[12];
  const float* AW1 = (const float*)d_in[13], *Ab1 = (const float*)d_in[14];
  const float* AW2 = (const float*)d_in[15], *Ab2 = (const float*)d_in[16];
  const float* BW1 = (const float*)d_in[17], *Bb1 = (const float*)d_in[18];
  const float* BW2 = (const float*)d_in[19], *Bb2 = (const float*)d_in[20];
  const float* CW1 = (const float*)d_in[21], *Cb1 = (const float*)d_in[22];
  const float* CW2 = (const float*)d_in[23], *Cb2 = (const float*)d_in[24];

  char* ws = (char*)d_ws;
  __bf16* wT   = (__bf16*)ws;                         // 5 * 128*128 bf16 = 163840 B
  float*  cs1  = (float*)(ws + 163840);               // 128 f32
  float*  degw = (float*)(ws + 164352);               // G*N f32 = 524288 B
  __bf16* x1   = (__bf16*)(ws + 688640);              // G*N*E bf16 = 33554432 B
  __bf16* x2   = (__bf16*)(ws + 688640 + 33554432);

  float* out = (float*)d_out;
  float* pA  = out;                 // [512,256]
  float* pB  = out + 131072;        // [512,256]
  float* pC  = out + 262144;        // [512,2]
  float* act = out + 263168;        // [512,3]

  dim3 lgrid(NN / 16, GG);
  k_prep<<<dim3(6), 256, 0, stream>>>(c1W1, c1W2, c2W1, c2W2, c3W1, c3W2, wT, cs1);
  k_layer1<<<lgrid, 256, 0, stream>>>(states, cs1, c1b1, wT + 0 * 16384, c1b2, x1, degw);
  k_layer <<<lgrid, 256, 0, stream>>>(states, x1, wT + 1 * 16384, c2b1, wT + 2 * 16384, c2b2, x2);
  k_layer <<<lgrid, 256, 0, stream>>>(states, x2, wT + 3 * 16384, c3b1, wT + 4 * 16384, c3b2, x1);

  size_t heads_smem = 65536 + (size_t)(128 * 32 + 256 * 32 + 128 * 32) * 4 +
                      (size_t)(256 + 256 + 128 + 256) * 4;   // 134656 B
  k_heads<<<GG, 256, heads_smem, stream>>>(x1, degw,
                                           AW1, Ab1, AW2, Ab2,
                                           BW1, Bb1, BW2, Bb2,
                                           CW1, Cb1, CW2, Cb2,
                                           pA, pB, pC, act);
}
